// InvNet_10660108829049
// MI455X (gfx1250) — compile-verified, matrix operand order checked by reading, not run on hardware
//
#include <hip/hip_runtime.h>

#define C_CLASSES 18530
#define FEAT 1280
#define BATCH 64
#define KNN 6
#define LDS_STRIDE 1296   // 1280 + 16 bf16 pad -> 8-bank stagger between M-rows

typedef __attribute__((ext_vector_type(16))) __bf16 v16bf;
typedef __attribute__((ext_vector_type(8)))  float  v8f;
typedef __attribute__((ext_vector_type(8)))  unsigned v8u;
typedef int v4i __attribute__((vector_size(16)));   // matches async-LDS builtin param type

#if defined(__has_builtin)
#if __has_builtin(__builtin_amdgcn_global_load_async_to_lds_b128)
#define HAVE_ASYNC_LDS 1
#endif
#if __has_builtin(__builtin_amdgcn_cvt_pk_bf16_f32)
#define HAVE_CVT_PK_BF16 1
#endif
#endif

__device__ __forceinline__ unsigned short f2bfu(float f) {
    unsigned u = __builtin_bit_cast(unsigned, f);
    u += 0x7FFFu + ((u >> 16) & 1u);          // round-to-nearest-even
    return (unsigned short)(u >> 16);
}
__device__ __forceinline__ __bf16 u2bf(unsigned short h) {
    return __builtin_bit_cast(__bf16, h);
}
// two f32 -> packed bf16x2 (dst[15:0]=lo, dst[31:16]=hi), hw cvt when available
__device__ __forceinline__ unsigned pack_bf16x2(float lo, float hi) {
#if HAVE_CVT_PK_BF16
    return __builtin_bit_cast(unsigned, __builtin_amdgcn_cvt_pk_bf16_f32(lo, hi));
#else
    return (unsigned)f2bfu(lo) | ((unsigned)f2bfu(hi) << 16);
#endif
}

// 16-byte global -> LDS copy, async (gfx1250 GLOBAL_LOAD_ASYNC_TO_LDS_B128) when available
__device__ __forceinline__ void copy16_g2l(const void* g, void* l) {
#if HAVE_ASYNC_LDS
    __builtin_amdgcn_global_load_async_to_lds_b128(
        (__attribute__((address_space(1))) v4i*)(unsigned long long)g,
        (__attribute__((address_space(3))) v4i*)(unsigned)(unsigned long long)l,
        0, 0);
#else
    *(uint4*)l = *(const uint4*)g;
#endif
}
__device__ __forceinline__ void wait_async_lds() {
#if HAVE_ASYNC_LDS
    asm volatile("s_wait_asynccnt 0" ::: "memory");
#endif
}

// ---------------------------------------------------------------- kernel 0
__global__ void __launch_bounds__(256)
cvt_kernel(const float* __restrict__ in, unsigned short* __restrict__ outp) {
    int i = blockIdx.x * 256 + threadIdx.x;
    if (i < BATCH * FEAT) outp[i] = f2bfu(in[i]);
}

// ---------------------------------------------------------------- kernel 1
// Fused: S = (inputs @ em^T) * rsqrt(||em_row||^2) / TEMP   (bf16 WMMA)
//        + em -> em_new bulk copy (em read exactly once from HBM).
// Full A (64x1280 bf16, ~162KB) staged in LDS ONCE -> barrier-free K loop.
// TAIL=false: all 128 classes of the block valid -> no exec masking at all.
// B stream is manually double-buffered (load kb+32 before consuming kb).
template <bool TAIL>
__global__ void __launch_bounds__(256)
gemm_norm_kernel(const unsigned short* __restrict__ inbf,
                 const float* __restrict__ em,
                 float* __restrict__ outEm,
                 float* __restrict__ S,
                 const int* __restrict__ epochPtr,
                 int blockOff) {
    __shared__ unsigned short ldsA[64 * LDS_STRIDE];  // ~162 KB of the WGP's 320 KB
    const int tid  = threadIdx.x;
    const int lane = tid & 31;
    const int wave = tid >> 5;                        // 8 waves
    const int half = (lane & 16);                     // 0 or 16 (K half)
    const int c    = (blockIdx.x + blockOff) * 128 + wave * 16 + (lane & 15);
    const bool valid = TAIL ? (c < C_CLASSES) : true;
    const float* emRow = em + (size_t)c * FEAT;

    // ---- one-shot A stage: 4 threads per M-row, 16B chunks ----
    {
        const int row = tid >> 2;                     // 0..63
        const int sub = (tid & 3) * 8;                // bf16 offset within 32-chunk
        for (int k = sub; k < FEAT; k += 32)
            copy16_g2l(inbf + row * FEAT + k, ldsA + row * LDS_STRIDE + k);
        wait_async_lds();
    }
    __syncthreads();

    v8f acc0 = {}, acc1 = {}, acc2 = {}, acc3 = {};
    float ss = 0.0f;
    const unsigned short* aRow = ldsA + (lane & 15) * LDS_STRIDE + half;

    auto loadB = [&](int kb, float4& g0, float4& g1, float4& g2, float4& g3) {
        if (!TAIL || valid) {
            g0 = *(const float4*)(emRow + kb + half + 0);
            g1 = *(const float4*)(emRow + kb + half + 4);
            g2 = *(const float4*)(emRow + kb + half + 8);
            g3 = *(const float4*)(emRow + kb + half + 12);
        } else {
            g0 = g1 = g2 = g3 = float4{0.f, 0.f, 0.f, 0.f};
        }
    };

    auto body = [&](int kb, float4 f0, float4 f1, float4 f2, float4 f3) {
        float t[16];
        t[0]=f0.x;  t[1]=f0.y;  t[2]=f0.z;  t[3]=f0.w;
        t[4]=f1.x;  t[5]=f1.y;  t[6]=f1.z;  t[7]=f1.w;
        t[8]=f2.x;  t[9]=f2.y;  t[10]=f2.z; t[11]=f2.w;
        t[12]=f3.x; t[13]=f3.y; t[14]=f3.z; t[15]=f3.w;
        v8u bu;
        #pragma unroll
        for (int i = 0; i < 8; i++) {
            ss += t[2*i] * t[2*i] + t[2*i+1] * t[2*i+1];
            bu[i] = pack_bf16x2(t[2*i], t[2*i+1]);    // v_cvt_pk_bf16_f32
        }
        v16bf bb = __builtin_bit_cast(v16bf, bu);
        if (!TAIL || valid) {  // fused em -> em_new copy (unaligned b128 stores ok)
            float* dst = outEm + (size_t)c * FEAT + kb + half;
            #pragma unroll
            for (int i = 0; i < 16; i++) dst[i] = t[i];
        }
        v16bf a0 = *(const v16bf*)(aRow + kb);
        v16bf a1 = *(const v16bf*)(aRow + 16 * LDS_STRIDE + kb);
        v16bf a2 = *(const v16bf*)(aRow + 32 * LDS_STRIDE + kb);
        v16bf a3 = *(const v16bf*)(aRow + 48 * LDS_STRIDE + kb);
        acc0 = __builtin_amdgcn_wmma_f32_16x16x32_bf16(false, a0, false, bb, (short)0, acc0, false, false);
        acc1 = __builtin_amdgcn_wmma_f32_16x16x32_bf16(false, a1, false, bb, (short)0, acc1, false, false);
        acc2 = __builtin_amdgcn_wmma_f32_16x16x32_bf16(false, a2, false, bb, (short)0, acc2, false, false);
        acc3 = __builtin_amdgcn_wmma_f32_16x16x32_bf16(false, a3, false, bb, (short)0, acc3, false, false);
    };

    float4 c0, c1, c2, c3;
    loadB(0, c0, c1, c2, c3);
    for (int kb = 0; kb < FEAT - 32; kb += 32) {
        float4 n0, n1, n2, n3;
        loadB(kb + 32, n0, n1, n2, n3);           // issue next chunk before use
        if (!TAIL && kb + 64 < FEAT)
            __builtin_prefetch(emRow + kb + 64 + half, 0, 1);
        body(kb, c0, c1, c2, c3);
        c0 = n0; c1 = n1; c2 = n2; c3 = n3;
    }
    body(FEAT - 32, c0, c1, c2, c3);              // peeled last iteration

    // fold the two K-halves of the per-class sum-of-squares
    float ssFull = ss + __shfl_xor(ss, 16, 32);
    const int epoch = *epochPtr;
    const float inv   = (epoch > 0) ? rsqrtf(ssFull) : 1.0f;
    const float scale = inv * 20.0f;                  // 1/TEMP

    if (valid) {
        const int mAdd = (lane & 16) ? 8 : 0;         // C layout: VGPR r -> M = r(+8)
        #pragma unroll
        for (int r = 0; r < 8; r++) {
            S[(size_t)( 0 + r + mAdd) * C_CLASSES + c] = acc0[r] * scale;
            S[(size_t)(16 + r + mAdd) * C_CLASSES + c] = acc1[r] * scale;
            S[(size_t)(32 + r + mAdd) * C_CLASSES + c] = acc2[r] * scale;
            S[(size_t)(48 + r + mAdd) * C_CLASSES + c] = acc3[r] * scale;
        }
    }
}

// ---------------------------------------------------------------- kernel 2
struct P { float v; int i; };
__device__ __forceinline__ bool betterP(P a, P b) {
    return (a.v > b.v) || (a.v == b.v && a.i < b.i);
}

__global__ void __launch_bounds__(256)
topk_kernel(const float* __restrict__ S, const int* __restrict__ domain,
            int* __restrict__ idxws) {
    __shared__ P sh[256 * KNN];
    const int b   = blockIdx.x;
    const int tid = threadIdx.x;
    const int db  = domain[b];
    const float* row = S + (size_t)b * C_CLASSES;

    P loc[KNN];
    #pragma unroll
    for (int k = 0; k < KNN; k++) { loc[k].v = -3.402823466e38f; loc[k].i = 0x7fffffff; }

    for (int c = tid; c < C_CLASSES; c += 256) {
        int dc = (c < 1816) ? 0 : (c < 3283) ? 1 : (c < 5095) ? 2 : (c < 6596) ? 3 : 4;
        bool keep = (db == 4) ? (dc == 4) : (dc != db);
        P cand; cand.v = keep ? row[c] : 0.0f; cand.i = c;   // feat = tgt * keep
        if (betterP(cand, loc[KNN - 1])) {
            loc[KNN - 1] = cand;
            #pragma unroll
            for (int t = KNN - 1; t > 0; t--)
                if (betterP(loc[t], loc[t - 1])) { P tmp = loc[t]; loc[t] = loc[t - 1]; loc[t - 1] = tmp; }
        }
    }
    #pragma unroll
    for (int k = 0; k < KNN; k++) sh[tid * KNN + k] = loc[k];
    __syncthreads();

    for (int s = 128; s > 0; s >>= 1) {
        if (tid < s) {
            P a0[KNN], b0[KNN], o[KNN];
            #pragma unroll
            for (int k = 0; k < KNN; k++) { a0[k] = sh[tid * KNN + k]; b0[k] = sh[(tid + s) * KNN + k]; }
            int ia = 0, ib = 0;
            #pragma unroll
            for (int k = 0; k < KNN; k++) {      // ia+ib==k<=5 -> never OOB
                if (betterP(a0[ia], b0[ib])) o[k] = a0[ia++]; else o[k] = b0[ib++];
            }
            #pragma unroll
            for (int k = 0; k < KNN; k++) sh[tid * KNN + k] = o[k];
        }
        __syncthreads();
    }
    if (tid == 0) {
        int id[KNN];
        #pragma unroll
        for (int k = 0; k < KNN; k++) id[k] = sh[k].i;
        for (int i = 0; i < KNN; i++)
            for (int j = i + 1; j < KNN; j++)
                if (id[j] < id[i]) { int t = id[i]; id[i] = id[j]; id[j] = t; }
        for (int k = 0; k < KNN; k++) idxws[b * KNN + k] = id[k];
    }
}

// ---------------------------------------------------------------- kernel 3
__device__ __forceinline__ float blockMax(float v, float* red) {
    int tid = threadIdx.x; red[tid] = v; __syncthreads();
    for (int s = 128; s > 0; s >>= 1) { if (tid < s) red[tid] = fmaxf(red[tid], red[tid + s]); __syncthreads(); }
    float r = red[0]; __syncthreads(); return r;
}
__device__ __forceinline__ float blockSum(float v, float* red) {
    int tid = threadIdx.x; red[tid] = v; __syncthreads();
    for (int s = 128; s > 0; s >>= 1) { if (tid < s) red[tid] += red[tid + s]; __syncthreads(); }
    float r = red[0]; __syncthreads(); return r;
}

__global__ void __launch_bounds__(256)
kl_kernel(const float* __restrict__ inputs, const float* __restrict__ em,
          const int* __restrict__ idxws, const int* __restrict__ epochPtr,
          float* __restrict__ acc) {
    __shared__ float red[256];
    const int j = blockIdx.x;
    const int b = j / KNN;
    const int ci = idxws[j];
    const float* x = inputs + (size_t)b * FEAT;
    const float* y = em + (size_t)ci * FEAT;
    const int tid = threadIdx.x;
    const int epoch = *epochPtr;
    float local = 0.0f;

    if (epoch > 0) {
        float mx = -3.402823466e38f, my = -3.402823466e38f;
        for (int f = tid; f < FEAT; f += 256) {
            mx = fmaxf(mx, x[f] * 20.0f);             // 1/BETA
            my = fmaxf(my, y[f] * 20.0f);
        }
        mx = blockMax(mx, red); my = blockMax(my, red);
        float sx = 0.0f, sy = 0.0f;
        for (int f = tid; f < FEAT; f += 256) {
            sx += __expf(x[f] * 20.0f - mx);
            sy += __expf(y[f] * 20.0f - my);
        }
        sx = blockSum(sx, red); sy = blockSum(sy, red);
        const float lsx = __logf(sx), lsy = __logf(sy);
        for (int f = tid; f < FEAT; f += 256) {
            float lxs = x[f] * 20.0f - mx, lys = y[f] * 20.0f - my;
            float p = __expf(lxs) / sx, q = __expf(lys) / sy;
            float lp = lxs - lsx, lq = lys - lsy;
            local += q * (lq - __logf(p + 1e-8f)) + p * (lp - __logf(q + 1e-8f));
        }
    } else {
        for (int f = tid; f < FEAT; f += 256) {
            float p = x[f], q = y[f];
            local += q * (__logf(q) - __logf(p + 1e-8f)) + p * (__logf(p) - __logf(q + 1e-8f));
        }
    }
    local = blockSum(local, red);
    if (tid == 0) atomicAdd(acc, local);
}

__global__ void finalize_kernel(const float* __restrict__ acc, float* __restrict__ out) {
    if (threadIdx.x == 0 && blockIdx.x == 0)
        out[0] = acc[0] * (1.0f / ((float)(BATCH * KNN) * (float)FEAT));
}

// ---------------------------------------------------------------- kernel 4
__global__ void __launch_bounds__(256)
copy_cur_kernel(const float* __restrict__ src, float* __restrict__ dst) {
    size_t i = (size_t)blockIdx.x * 256 + threadIdx.x;
    const size_t stride = (size_t)gridDim.x * 256;
    const size_t n = (size_t)C_CLASSES * FEAT;
    for (; i < n; i += stride) dst[i] = src[i];
}

__global__ void __launch_bounds__(256)
scatter_kernel(const float* __restrict__ inputs, const int* __restrict__ label,
               const float* __restrict__ em_last, const float* __restrict__ em_cur,
               const float* __restrict__ em_cnt, const int* __restrict__ epochPtr,
               float* __restrict__ outEm, float* __restrict__ outCur) {
    const int b = blockIdx.x;
    const int y = label[b];
    const float n  = em_cnt[y] + 1.0f;
    const float c1 = (n - 1.0f) / n, c2 = 1.0f / n;
    const float alpha = (*epochPtr == 0) ? 0.0f : 0.01f;
    for (int f = threadIdx.x; f < FEAT; f += 256) {
        float u = em_cur[(size_t)y * FEAT + f] * c1 + inputs[(size_t)b * FEAT + f] * c2;
        outCur[(size_t)y * FEAT + f] = u;
        outEm[(size_t)y * FEAT + f]  = em_last[(size_t)y * FEAT + f] * alpha + u * (1.0f - alpha);
    }
}

// ---------------------------------------------------------------- launcher
extern "C" void kernel_launch(void* const* d_in, const int* in_sizes, int n_in,
                              void* d_out, int out_size, void* d_ws, size_t ws_size,
                              hipStream_t stream) {
    const float* inputs  = (const float*)d_in[0];
    const int*   label   = (const int*)d_in[1];
    const int*   domain  = (const int*)d_in[2];
    const float* em      = (const float*)d_in[3];
    const float* em_last = (const float*)d_in[4];
    const float* em_cur  = (const float*)d_in[5];
    const float* em_cnt  = (const float*)d_in[6];
    const int*   epoch   = (const int*)d_in[7];

    float* out    = (float*)d_out;
    float* outEm  = out + 1;
    float* outCur = out + 1 + (size_t)C_CLASSES * FEAT;

    char* ws = (char*)d_ws;
    unsigned short* inbf = (unsigned short*)ws;                       // 163,840 B
    float* S   = (float*)(ws + 163840);                               // 4,743,680 B
    int*   idx = (int*)(ws + 163840 + 4743680);                       // 1,536 B
    float* acc = (float*)(ws + 163840 + 4743680 + 1536);              // 4 B

    (void)hipMemsetAsync(acc, 0, sizeof(float), stream);

    const int FULL_BLOCKS = C_CLASSES / 128;                          // 144 guard-free blocks
    cvt_kernel<<<(BATCH * FEAT + 255) / 256, 256, 0, stream>>>(inputs, inbf);
    gemm_norm_kernel<false><<<FULL_BLOCKS, 256, 0, stream>>>(inbf, em, outEm, S, epoch, 0);
    gemm_norm_kernel<true ><<<1, 256, 0, stream>>>(inbf, em, outEm, S, epoch, FULL_BLOCKS);
    topk_kernel<<<BATCH, 256, 0, stream>>>(S, domain, idx);
    kl_kernel<<<BATCH * KNN, 256, 0, stream>>>(inputs, em, idx, epoch, acc);
    finalize_kernel<<<1, 32, 0, stream>>>(acc, out);
    copy_cur_kernel<<<2048, 256, 0, stream>>>(em_cur, outCur);
    scatter_kernel<<<BATCH, 256, 0, stream>>>(inputs, label, em_last, em_cur, em_cnt,
                                              epoch, outEm, outCur);
}
